// KGNN_41566693491231
// MI455X (gfx1250) — matrix operand order repaired and checked by reading, not compiled, not run on hardware
//
#include <hip/hip_runtime.h>
#include <hip/hip_bf16.h>

typedef __bf16 bf16;
typedef __attribute__((ext_vector_type(16))) bf16  v16bf;
typedef __attribute__((ext_vector_type(8)))  bf16  v8bf;
typedef __attribute__((ext_vector_type(4)))  bf16  v4bf;
typedef __attribute__((ext_vector_type(8)))  float v8f;

#define KG_N_NODES 100000
#define KG_N_EDGES 600000
#define KG_H 128

union FragU { v16bf v; v8bf h[2]; };

// A-fragment: 16-bit A 16x32 layout (ISA 7.12.2): lane half selects K base
// {0 or 8}; elements 0..7 = K kb..kb+7, elements 8..15 = K kb+16..kb+23.
static __device__ __forceinline__ v16bf load_a_frag(const bf16* p, int kb) {
  FragU u;
  u.h[0] = *(const v8bf*)(p + kb);
  u.h[1] = *(const v8bf*)(p + kb + 16);
  return u.v;
}
// B-fragment: same K layout as A: elements 0..7 = K kb..kb+7, 8..15 = K kb+16..kb+23.
static __device__ __forceinline__ v16bf load_b_frag(const bf16* p, int kb) {
  FragU u;
  u.h[0] = *(const v8bf*)(p + kb);
  u.h[1] = *(const v8bf*)(p + kb + 16);
  return u.v;
}

// ---------------------------------------------------------------- utilities
__global__ void zero_f4(float4* __restrict__ p, int n4) {
  int i = blockIdx.x * blockDim.x + threadIdx.x;
  if (i < n4) p[i] = make_float4(0.f, 0.f, 0.f, 0.f);
}

// WT[n*K + k] = (n < Nsrc) ? bf16(W[k*Nsrc + n]) : 0   (W stored [K,Nsrc])
__global__ void convert_wt(const float* __restrict__ W, bf16* __restrict__ WT,
                           int K, int Nsrc, int Npad) {
  int i = blockIdx.x * blockDim.x + threadIdx.x;
  if (i >= Npad * K) return;
  int n = i / K, k = i - n * K;
  WT[i] = (n < Nsrc) ? (bf16)W[k * Nsrc + n] : (bf16)0.0f;
}

// e50 = relu(l2norm(rel_table) @ lin_W + lin_b)  (50 x 128, trivial FLOPs)
__global__ void prep_rel(const float* __restrict__ relT, const float* __restrict__ linW,
                         const float* __restrict__ linb, float* __restrict__ e50) {
  int idx = blockIdx.x * blockDim.x + threadIdx.x;
  if (idx >= 50 * KG_H) return;
  int r = idx >> 7, j = idx & (KG_H - 1);
  const float* row = relT + r * KG_H;
  float ss = 0.f;
  for (int k = 0; k < KG_H; ++k) ss += row[k] * row[k];
  float scale = 1.0f / fmaxf(sqrtf(ss), 1e-12f);
  float acc = linb[j];
  for (int k = 0; k < KG_H; ++k) acc += (row[k] * scale) * linW[k * KG_H + j];
  e50[idx] = fmaxf(acc, 0.0f);
}

// Ab[row] = bf16(l2norm(node_emb_table[node_ids[row]]))  (one wave32 per row)
__global__ void prep_nodes(const float* __restrict__ table, const int* __restrict__ ids,
                           bf16* __restrict__ out, int n) {
  int row = blockIdx.x * (blockDim.x >> 5) + (threadIdx.x >> 5);
  if (row >= n) return;
  int lane = threadIdx.x & 31;
  const float4* p = (const float4*)(table + (size_t)ids[row] * KG_H);
  float4 v = p[lane];
  float ss = v.x * v.x + v.y * v.y + v.z * v.z + v.w * v.w;
#pragma unroll
  for (int o = 16; o >= 1; o >>= 1) ss += __shfl_xor(ss, o, 32);
  float scale = 1.0f / fmaxf(sqrtf(ss), 1e-12f);
  v4bf q = { (bf16)(v.x * scale), (bf16)(v.y * scale),
             (bf16)(v.z * scale), (bf16)(v.w * scale) };
  *(v4bf*)(out + (size_t)row * KG_H + lane * 4) = q;
}

// agg[dst] += relu(x[src] + e50[rel])   (one wave32 per edge, 4 floats/lane)
__global__ void gine_scatter(const float* __restrict__ x, const float* __restrict__ e50,
                             const int* __restrict__ src, const int* __restrict__ dst,
                             const int* __restrict__ rel, float* __restrict__ agg, int E) {
  int e = blockIdx.x * (blockDim.x >> 5) + (threadIdx.x >> 5);
  if (e >= E) return;
  int lane = threadIdx.x & 31;
  int s = src[e], d = dst[e], r = rel[e];
  float4 xs = *(const float4*)(x + (size_t)s * KG_H + lane * 4);
  float4 es = *(const float4*)(e50 + (size_t)r * KG_H + lane * 4);
  float* a = agg + (size_t)d * KG_H + lane * 4;
  atomicAdd(a + 0, fmaxf(xs.x + es.x, 0.f));
  atomicAdd(a + 1, fmaxf(xs.y + es.y, 0.f));
  atomicAdd(a + 2, fmaxf(xs.z + es.z, 0.f));
  atomicAdd(a + 3, fmaxf(xs.w + es.w, 0.f));
}

// Ab = bf16(x + agg)
__global__ void combine_xagg(const float* __restrict__ x, const float* __restrict__ agg,
                             bf16* __restrict__ Ab, int n4) {
  int i = blockIdx.x * blockDim.x + threadIdx.x;
  if (i >= n4) return;
  float4 a = ((const float4*)x)[i];
  float4 g = ((const float4*)agg)[i];
  v4bf q = { (bf16)(a.x + g.x), (bf16)(a.y + g.y),
             (bf16)(a.z + g.z), (bf16)(a.w + g.w) };
  *(v4bf*)(Ab + (size_t)i * 4) = q;
}

// ------------------------------------------------------ generic WMMA GEMM
// out[m][n] = (relu?) A[rowids[m]] (bf16,[*,K]) @ WT^T (WT is [Npad][K] bf16)
// + bias[n].  One wave per 16-row tile, loops over ntiles of 16 columns.
template <int KTILES, bool RELU, bool WRITE_B16>
__global__ __launch_bounds__(256)
void wmma_gemm_rows(const bf16* __restrict__ A, const bf16* __restrict__ WT,
                    const float* __restrict__ bias, float* __restrict__ out,
                    bf16* outb, const int* __restrict__ rowids,
                    int M, int Nout, int ntiles, int ldout) {
  constexpr int K = KTILES * 32;
  int wave = blockIdx.x * (blockDim.x >> 5) + (threadIdx.x >> 5);
  int mbase = wave * 16;
  if (mbase >= M) return;
  int lane = threadIdx.x & 31;
  int hl = lane >> 4;           // half of the wave
  int ml = lane & 15;
  int row = mbase + ml;
  if (row >= M) row = M - 1;
  int rid = rowids ? rowids[row] : row;
  const bf16* ap = A + (size_t)rid * K;

  v16bf afrag[KTILES];
#pragma unroll
  for (int ks = 0; ks < KTILES; ++ks)
    afrag[ks] = load_a_frag(ap, ks * 32 + hl * 8);

  for (int nt = 0; nt < ntiles; ++nt) {
    int n = nt * 16 + ml;
    float bv = (n < Nout) ? bias[n] : 0.f;
    v8f c = { bv, bv, bv, bv, bv, bv, bv, bv };
    const bf16* bp = WT + (size_t)n * K;
#pragma unroll
    for (int ks = 0; ks < KTILES; ++ks) {
      v16bf bfrag = load_b_frag(bp, ks * 32 + hl * 8);
      c = __builtin_amdgcn_wmma_f32_16x16x32_bf16(
              false, afrag[ks], false, bfrag, (short)0, c, false, false);
    }
    if (n < Nout) {
      int mrow = mbase + hl * 8;
#pragma unroll
      for (int r2 = 0; r2 < 8; ++r2) {
        int mm = mrow + r2;
        if (mm < M) {
          float v = c[r2];
          if (RELU) v = fmaxf(v, 0.f);
          out[(size_t)mm * ldout + n] = v;
          if (WRITE_B16) outb[(size_t)mm * ldout + n] = (bf16)v;
        }
      }
    }
  }
}

// Edge classifier: A = concat(xb[src[e]], xb[dst[e]]) (K=256), N=50 (pad 64).
__global__ __launch_bounds__(256)
void wmma_gemm_edges(const bf16* __restrict__ xb, const int* __restrict__ src,
                     const int* __restrict__ dst, const bf16* __restrict__ WT,
                     const float* __restrict__ bias, float* __restrict__ out, int E) {
  int wave = blockIdx.x * (blockDim.x >> 5) + (threadIdx.x >> 5);
  int ebase = wave * 16;
  if (ebase >= E) return;
  int lane = threadIdx.x & 31;
  int hl = lane >> 4;
  int ml = lane & 15;
  int e = ebase + ml;
  if (e >= E) e = E - 1;
  const bf16* ps = xb + (size_t)src[e] * KG_H;
  const bf16* pd = xb + (size_t)dst[e] * KG_H;

  v16bf afrag[8];
#pragma unroll
  for (int ks = 0; ks < 8; ++ks) {
    const bf16* p = (ks < 4) ? ps : pd;           // chunks never straddle K=128
    afrag[ks] = load_a_frag(p, (ks & 3) * 32 + hl * 8);
  }

  for (int nt = 0; nt < 4; ++nt) {
    int n = nt * 16 + ml;
    float bv = (n < 50) ? bias[n] : 0.f;
    v8f c = { bv, bv, bv, bv, bv, bv, bv, bv };
    const bf16* bp = WT + (size_t)n * 256;
#pragma unroll
    for (int ks = 0; ks < 8; ++ks) {
      v16bf bfrag = load_b_frag(bp, ks * 32 + hl * 8);
      c = __builtin_amdgcn_wmma_f32_16x16x32_bf16(
              false, afrag[ks], false, bfrag, (short)0, c, false, false);
    }
    if (n < 50) {
      int eb = ebase + hl * 8;
#pragma unroll
      for (int r2 = 0; r2 < 8; ++r2) {
        int ee = eb + r2;
        if (ee < E) out[(size_t)ee * 50 + n] = c[r2];
      }
    }
  }
}

// ---------------------------------------------------------------- launcher
extern "C" void kernel_launch(void* const* d_in, const int* in_sizes, int n_in,
                              void* d_out, int out_size, void* d_ws, size_t ws_size,
                              hipStream_t stream) {
  const int*   node_ids = (const int*)d_in[0];
  const int*   rel_ids  = (const int*)d_in[1];
  const int*   center   = (const int*)d_in[2];
  const int*   nonmol   = (const int*)d_in[3];
  const int*   eidx     = (const int*)d_in[4];
  const float* node_tab = (const float*)d_in[5];
  const float* rel_tab  = (const float*)d_in[6];
  const float* linW  = (const float*)d_in[7];
  const float* linb  = (const float*)d_in[8];
  const float* W1    = (const float*)d_in[9];
  const float* b1    = (const float*)d_in[10];
  const float* W2    = (const float*)d_in[11];
  const float* b2    = (const float*)d_in[12];
  const float* edgeW = (const float*)d_in[13];
  const float* edgeb = (const float*)d_in[14];
  const float* motifW = (const float*)d_in[15];
  const float* motifb = (const float*)d_in[16];
  const float* nodecW = (const float*)d_in[17];
  const float* nodecb = (const float*)d_in[18];
  const float* binW  = (const float*)d_in[19];
  const float* binb  = (const float*)d_in[20];

  char* ws = (char*)d_ws;
  float* x      = (float*)(ws +          0);   // 100000*128*4 = 51,200,000
  float* agg    = (float*)(ws +  51200000);    // 51,200,000
  bf16*  Ab     = (bf16*) (ws + 102400000);    // 25,600,000
  bf16*  xb     = (bf16*) (ws + 128000000);    // 25,600,000
  float* e50    = (float*)(ws + 153600000);    // 25,600
  bf16*  linT   = (bf16*) (ws + 153625600);    // 128x128 bf16
  bf16*  W1T    = (bf16*) (ws + 153658368);
  bf16*  W2T    = (bf16*) (ws + 153691136);
  bf16*  edgeT  = (bf16*) (ws + 153723904);    // 64x256
  bf16*  motifT = (bf16*) (ws + 153756672);    // 96x128
  bf16*  nodecT = (bf16*) (ws + 153781248);    // 16x128
  bf16*  binT   = (bf16*) (ws + 153785344);    // 16x128

  const int* src = eidx;
  const int* dst = eidx + KG_N_EDGES;
  float* out_edge  = (float*)d_out;                          // 600000*50
  float* out_motif = out_edge + (size_t)KG_N_EDGES * 50;     // 512*85
  float* out_node  = out_motif + 512 * 85;                   // 20000*15
  float* out_bin   = out_node + 20000 * 15;                  // 100000*1

  auto cdiv = [](int a, int b) { return (a + b - 1) / b; };
  const int T = 256;

  // 1) weights -> bf16, [Npad][K] transposed layout
  convert_wt<<<cdiv(128 * 128, T), T, 0, stream>>>(linW,   linT,   128, 128, 128);
  convert_wt<<<cdiv(128 * 128, T), T, 0, stream>>>(W1,     W1T,    128, 128, 128);
  convert_wt<<<cdiv(128 * 128, T), T, 0, stream>>>(W2,     W2T,    128, 128, 128);
  convert_wt<<<cdiv(64 * 256,  T), T, 0, stream>>>(edgeW,  edgeT,  256,  50,  64);
  convert_wt<<<cdiv(96 * 128,  T), T, 0, stream>>>(motifW, motifT, 128,  85,  96);
  convert_wt<<<cdiv(16 * 128,  T), T, 0, stream>>>(nodecW, nodecT, 128,  15,  16);
  convert_wt<<<cdiv(16 * 128,  T), T, 0, stream>>>(binW,   binT,   128,   1,  16);

  // 2) 50 distinct relation features (dedups the 600k-row edge GEMM)
  prep_rel<<<cdiv(50 * KG_H, T), T, 0, stream>>>(rel_tab, linW, linb, e50);

  // 3) gather + l2norm node embeddings -> bf16
  prep_nodes<<<KG_N_NODES / 8, T, 0, stream>>>(node_tab, node_ids, Ab, KG_N_NODES);

  const int n4     = KG_N_NODES * KG_H / 4;          // float4 count
  const int gbRows = cdiv(KG_N_NODES / 16, 8);       // wave-per-16-rows blocks

  // 4) x = relu(Ab @ lin_W + lin_b)
  wmma_gemm_rows<4, true, false><<<gbRows, T, 0, stream>>>(
      Ab, linT, linb, x, nullptr, nullptr, KG_N_NODES, 128, 8, 128);

  // 5) GINE layer 1
  zero_f4<<<cdiv(n4, T), T, 0, stream>>>((float4*)agg, n4);
  gine_scatter<<<KG_N_EDGES / 8, T, 0, stream>>>(x, e50, src, dst, rel_ids, agg, KG_N_EDGES);
  combine_xagg<<<cdiv(n4, T), T, 0, stream>>>(x, agg, Ab, n4);
  wmma_gemm_rows<4, true, false><<<gbRows, T, 0, stream>>>(
      Ab, W1T, b1, x, nullptr, nullptr, KG_N_NODES, 128, 8, 128);

  // 6) GINE layer 2 (no relu) -> x fp32 + xb bf16
  zero_f4<<<cdiv(n4, T), T, 0, stream>>>((float4*)agg, n4);
  gine_scatter<<<KG_N_EDGES / 8, T, 0, stream>>>(x, e50, src, dst, rel_ids, agg, KG_N_EDGES);
  combine_xagg<<<cdiv(n4, T), T, 0, stream>>>(x, agg, Ab, n4);
  wmma_gemm_rows<4, false, true><<<gbRows, T, 0, stream>>>(
      Ab, W2T, b2, x, xb, nullptr, KG_N_NODES, 128, 8, 128);

  // 7) heads
  wmma_gemm_edges<<<cdiv(KG_N_EDGES / 16, 8), T, 0, stream>>>(
      xb, src, dst, edgeT, edgeb, out_edge, KG_N_EDGES);
  wmma_gemm_rows<4, false, false><<<cdiv(512 / 16, 8), T, 0, stream>>>(
      xb, motifT, motifb, out_motif, nullptr, center, 512, 85, 6, 85);
  wmma_gemm_rows<4, false, false><<<cdiv(20000 / 16, 8), T, 0, stream>>>(
      xb, nodecT, nodecb, out_node, nullptr, nonmol, 20000, 15, 1, 15);
  wmma_gemm_rows<4, false, false><<<gbRows, T, 0, stream>>>(
      xb, binT, binb, out_bin, nullptr, nullptr, KG_N_NODES, 1, 1, 1);

  (void)in_sizes; (void)n_in; (void)out_size; (void)ws_size;
}
